// DTMPLayer_15779709846143
// MI455X (gfx1250) — matrix operand order repaired and checked by reading, not compiled, not run on hardware
//
#include <hip/hip_runtime.h>

#define DN   100
#define DE   172
#define DT   100
#define DOUT 128
#define NH   2
#define HD   64
#define KV_KIN 372
#define KV_K   384
#define Q_KIN  200
#define Q_K    224
#define O_KIN  228
#define O_K    256

typedef __bf16 bf16;
typedef __attribute__((ext_vector_type(16))) __bf16 v16bf;
typedef __attribute__((ext_vector_type(8)))  __bf16 v8bf;
typedef __attribute__((ext_vector_type(8)))  float  v8f;
typedef __attribute__((ext_vector_type(4)))  float  v4f;

// ---- monotone float <-> uint mapping for atomicMax-based segmented max ----
__device__ __forceinline__ unsigned f32_sortable(float f) {
  unsigned u = __float_as_uint(f);
  return (u & 0x80000000u) ? ~u : (u | 0x80000000u);
}
__device__ __forceinline__ float sortable_f32(unsigned u) {
  return (u & 0x80000000u) ? __uint_as_float(u & 0x7fffffffu)
                           : __uint_as_float(~u);
}

// ---- WMMA helpers -------------------------------------------------------
__device__ __forceinline__ v8f wmma_bf16(v16bf a, v16bf b, v8f c) {
  // D = A(16x32 bf16) * B(32x16 bf16) + C(16x16 f32)
  return __builtin_amdgcn_wmma_f32_16x16x32_bf16(
      false, a, false, b, (short)0, c, false, false);
}

// A fragment for 16-bit A 16x32 layout (ISA 7.12.2):
//  lanes 0-15 : row M=lane,    K = k0+0..7   and k0+16..23
//  lanes16-31 : row M=lane-16, K = k0+8..15  and k0+24..31
__device__ __forceinline__ v16bf load_a_frag(const bf16* A, int kpad, int lane, int ch) {
  int row = lane & 15;
  int hi  = lane >> 4;
  const v8bf* p = (const v8bf*)&A[row * kpad + ch * 32 + hi * 8];
  v8bf lo = p[0];
  v8bf up = p[2];
  v16bf a;
#pragma unroll
  for (int j = 0; j < 8; ++j) { a[j] = lo[j]; a[8 + j] = up[j]; }
  return a;
}

// B fragment from pre-swizzled weights: each lane's 16 bf16 stored contiguously
// at [((wid*nch + ch)*32 + lane)*16] -> two global_load_b128 per fragment.
__device__ __forceinline__ v16bf load_b_frag(const bf16* __restrict__ S, int nch,
                                             int lane, int wid, int ch) {
  const v8bf* p = (const v8bf*)&S[(((wid * nch) + ch) * 32 + lane) * 16];
  v8bf lo = p[0];
  v8bf up = p[1];
  v16bf b;
#pragma unroll
  for (int j = 0; j < 8; ++j) { b[j] = lo[j]; b[8 + j] = up[j]; }
  return b;
}

// ---- weight pre-swizzle: W[k][n] f32 -> fragment-ready bf16 -------------
__device__ __forceinline__ void swizzle_w(const float* __restrict__ W, bf16* __restrict__ S,
                                          int nch, int kin, int i) {
  // i = ((wid*nch + ch)*32 + lane)*16 + j
  int wid  = i / (nch * 512);
  int rem  = i - wid * nch * 512;
  int ch   = rem >> 9;
  int r2   = rem & 511;
  int lane = r2 >> 4;
  int j    = r2 & 15;
  int col  = wid * 16 + (lane & 15);
  int k    = ch * 32 + (lane >> 4) * 8 + (j & 7) + (j >> 3) * 16;
  S[i] = (k < kin) ? (bf16)W[k * DOUT + col] : (bf16)0.f;
}

__global__ void k_prep_weights(const float* __restrict__ wq, const float* __restrict__ wk,
                               const float* __restrict__ wv, const float* __restrict__ wo,
                               bf16* __restrict__ wqs, bf16* __restrict__ wks,
                               bf16* __restrict__ wvs, bf16* __restrict__ wos) {
  int i = blockIdx.x * blockDim.x + threadIdx.x;
  if (i < 8 * (KV_K / 32) * 512) {
    swizzle_w(wk, wks, KV_K / 32, KV_KIN, i);
    swizzle_w(wv, wvs, KV_K / 32, KV_KIN, i);
  }
  if (i < 8 * (Q_K / 32) * 512) swizzle_w(wq, wqs, Q_K / 32, Q_KIN, i);
  if (i < 8 * (O_K / 32) * 512) swizzle_w(wo, wos, O_K / 32, O_KIN, i);
}

// ---- zero/seed accumulators (fresh every call) --------------------------
__global__ void k_init(unsigned* __restrict__ m_u, float* __restrict__ z,
                       float* __restrict__ agg, int num_dst) {
  int i = blockIdx.x * blockDim.x + threadIdx.x;
  if (i < num_dst * DOUT) agg[i] = 0.f;
  if (i < num_dst * NH) { m_u[i] = 0u; z[i] = 0.f; }
}

// ---- vectorized staging of one 16-edge kv_in tile into LDS (bf16) -------
__device__ __forceinline__ void stage_kv_tile(bf16* __restrict__ A,
    const float* __restrict__ h, const float* __restrict__ ef,
    const float* __restrict__ dtv, const float* __restrict__ time_w,
    const float* __restrict__ time_b, int t, int E, int num_dst, int tid) {
  for (int run = tid; run < 16 * (KV_K / 8); run += 256) {
    int row = run / (KV_K / 8);
    int k0  = (run - row * (KV_K / 8)) * 8;
    int e   = t * 16 + row;
    float v[8];
    if (e >= E) {
#pragma unroll
      for (int j = 0; j < 8; ++j) v[j] = 0.f;
    } else if (k0 + 8 <= DN) {                       // fully node features
      const v4f* p = (const v4f*)&h[(long)(num_dst + e) * DN + k0];
      v4f a0 = p[0], a1 = p[1];
#pragma unroll
      for (int j = 0; j < 4; ++j) { v[j] = a0[j]; v[4 + j] = a1[j]; }
    } else if (k0 >= DN + 4 && k0 + 8 <= DN + DE) {  // fully edge features
      const v4f* p = (const v4f*)&ef[(long)e * DE + (k0 - DN)];
      v4f a0 = p[0], a1 = p[1];
#pragma unroll
      for (int j = 0; j < 4; ++j) { v[j] = a0[j]; v[4 + j] = a1[j]; }
    } else if (k0 >= DN + DE && k0 + 8 <= KV_KIN) {  // fully time encoding
      float d = dtv[e];
      int ti = k0 - (DN + DE);
#pragma unroll
      for (int j = 0; j < 8; ++j) v[j] = __cosf(d * time_w[ti + j] + time_b[ti + j]);
    } else {                                         // straddling runs (k0=96,368,376)
#pragma unroll
      for (int j = 0; j < 8; ++j) {
        int k = k0 + j;
        float x = 0.f;
        if (k < DN)           x = h[(long)(num_dst + e) * DN + k];
        else if (k < DN + DE) x = ef[(long)e * DE + (k - DN)];
        else if (k < KV_KIN)  { int ti = k - (DN + DE);
                                x = __cosf(dtv[e] * time_w[ti] + time_b[ti]); }
        v[j] = x;
      }
    }
    v8bf b;
#pragma unroll
    for (int j = 0; j < 8; ++j) b[j] = (bf16)v[j];
    *(v8bf*)&A[row * KV_K + k0] = b;                 // one ds_store_b128
  }
}

// prefetch next tile's feature rows (global_prefetch_b8)
__device__ __forceinline__ void prefetch_tile(const float* __restrict__ h,
    const float* __restrict__ ef, int tnext, int ntiles, int E, int num_dst, int tid) {
  if (tnext >= ntiles) return;
  if (tid < 64) {                                    // h rows: 400B -> 4 lines
    int row = tid >> 2, seg = tid & 3;
    long e = (long)tnext * 16 + row;
    if (e < E) __builtin_prefetch(&h[(num_dst + e) * DN + seg * 32], 0, 0);
  } else if (tid < 160) {                            // ef rows: 688B -> 6 lines
    int r = tid - 64;
    int row = r / 6, seg = r - row * 6;
    long e = (long)tnext * 16 + row;
    if (e < E) __builtin_prefetch(&ef[e * DE + seg * 32], 0, 0);
  }
}

// ---- Q projection per dst node ------------------------------------------
__global__ void __launch_bounds__(256)
k_qnodes(const float* __restrict__ h, const float* __restrict__ time_b,
         const bf16* __restrict__ wqs, const float* __restrict__ bq,
         float* __restrict__ q_nodes, int num_dst) {
  __shared__ bf16 A[16 * Q_K];
  int tile = blockIdx.x, tid = threadIdx.x;
  for (int idx = tid; idx < 16 * Q_K; idx += 256) {
    int row = idx / Q_K, k = idx - row * Q_K;
    int d = tile * 16 + row;
    float v = 0.f;
    if (d < num_dst) {
      if (k < DN)           v = h[(long)d * DN + k];
      else if (k < DN + DT) v = __cosf(time_b[k - DN]);   // zero-time encoding
    }
    A[idx] = (bf16)v;
  }
  __syncthreads();
  int lane = tid & 31, wid = tid >> 5;
  int col = wid * 16 + (lane & 15);
  int hi  = lane >> 4;
  v8f c = {};
#pragma unroll
  for (int ch = 0; ch < Q_K / 32; ++ch) {
    v16bf a = load_a_frag(A, Q_K, lane, ch);
    v16bf b = load_b_frag(wqs, Q_K / 32, lane, wid, ch);
    c = wmma_bf16(a, b, c);
  }
  float bqc = bq[col];
#pragma unroll
  for (int r = 0; r < 8; ++r) {
    int d = tile * 16 + r + hi * 8;
    if (d < num_dst) q_nodes[(long)d * DOUT + col] = c[r] + bqc;
  }
}

// ---- K projection + attention scores + segmented max --------------------
__global__ void __launch_bounds__(256)
k_kv_score(const float* __restrict__ h, const float* __restrict__ ef,
           const float* __restrict__ dtv, const int* __restrict__ dst_idx,
           const float* __restrict__ time_w, const float* __restrict__ time_b,
           const bf16* __restrict__ wks, const float* __restrict__ bk,
           const float* __restrict__ att_bias, const float* __restrict__ q_nodes,
           float* __restrict__ scores, unsigned* __restrict__ m_u,
           int E, int num_dst) {
  __shared__ bf16  A[16 * KV_K];
  __shared__ float Kt[16][DOUT];
  __shared__ float bias_sum[NH];
  int tid = threadIdx.x, lane = tid & 31, wid = tid >> 5;
  int col = wid * 16 + (lane & 15);
  int hi  = lane >> 4;

  v16bf bfrag[KV_K / 32];                 // weights resident in VGPRs
#pragma unroll
  for (int ch = 0; ch < KV_K / 32; ++ch)
    bfrag[ch] = load_b_frag(wks, KV_K / 32, lane, wid, ch);

  if (tid < NH) {
    float s = 0.f;
    for (int j = 0; j < HD; ++j) s += att_bias[tid * HD + j];
    bias_sum[tid] = s;
  }
  float bkc = bk[col];
  int ntiles = (E + 15) >> 4;

  for (int t = blockIdx.x; t < ntiles; t += gridDim.x) {
    stage_kv_tile(A, h, ef, dtv, time_w, time_b, t, E, num_dst, tid);
    prefetch_tile(h, ef, t + gridDim.x, ntiles, E, num_dst, tid);
    __syncthreads();
    v8f c = {};
#pragma unroll
    for (int ch = 0; ch < KV_K / 32; ++ch) {
      v16bf a = load_a_frag(A, KV_K, lane, ch);
      c = wmma_bf16(a, bfrag[ch], c);
    }
#pragma unroll
    for (int r = 0; r < 8; ++r) Kt[r + hi * 8][col] = c[r] + bkc;
    __syncthreads();
    if (tid < 32) {                       // one (edge, head) pair per lane
      int m = tid >> 1, hh = tid & 1;
      int e = t * 16 + m;
      if (e < E) {
        int d = dst_idx[e];
        const float* q = &q_nodes[(long)d * DOUT + hh * HD];
        float s = bias_sum[hh];
#pragma unroll 8
        for (int j = 0; j < HD; ++j) s += q[j] * Kt[m][hh * HD + j];
        s = (s > 0.f) ? s : 0.2f * s;     // leaky_relu(0.2)
        scores[(long)e * NH + hh] = s;
        atomicMax(&m_u[d * NH + hh], f32_sortable(s));
      }
    }
    __syncthreads();
  }
}

// ---- exp(s - max) + segmented sum ---------------------------------------
__global__ void k_exp(const float* __restrict__ scores, const int* __restrict__ dst_idx,
                      const unsigned* __restrict__ m_u, float* __restrict__ ev,
                      float* __restrict__ z, int E) {
  int i = blockIdx.x * blockDim.x + threadIdx.x;
  if (i >= E * NH) return;
  int e = i >> 1, hh = i & 1;
  int d = dst_idx[e];
  float mu = sortable_f32(m_u[d * NH + hh]);
  float x = __expf(scores[i] - mu);
  ev[i] = x;
  atomicAdd(&z[d * NH + hh], x);
}

// ---- V projection + att-weighted scatter-add ----------------------------
__global__ void __launch_bounds__(256)
k_v_agg(const float* __restrict__ h, const float* __restrict__ ef,
        const float* __restrict__ dtv, const int* __restrict__ dst_idx,
        const float* __restrict__ time_w, const float* __restrict__ time_b,
        const bf16* __restrict__ wvs, const float* __restrict__ bv,
        const float* __restrict__ ev, const float* __restrict__ z,
        float* __restrict__ agg, int E, int num_dst) {
  __shared__ bf16  A[16 * KV_K];
  __shared__ float att[16][NH];
  __shared__ int   dstr[16];
  int tid = threadIdx.x, lane = tid & 31, wid = tid >> 5;
  int col = wid * 16 + (lane & 15);
  int hi  = lane >> 4;
  int head = col >> 6;

  v16bf bfrag[KV_K / 32];
#pragma unroll
  for (int ch = 0; ch < KV_K / 32; ++ch)
    bfrag[ch] = load_b_frag(wvs, KV_K / 32, lane, wid, ch);
  float bvc = bv[col];
  int ntiles = (E + 15) >> 4;

  for (int t = blockIdx.x; t < ntiles; t += gridDim.x) {
    stage_kv_tile(A, h, ef, dtv, time_w, time_b, t, E, num_dst, tid);
    prefetch_tile(h, ef, t + gridDim.x, ntiles, E, num_dst, tid);
    if (tid >= 160 && tid < 192) {        // att/dst staging on spare lanes
      int r = tid - 160;
      int m = r >> 1, hh = r & 1;
      int e = t * 16 + m;
      if (e < E) {
        int d = dst_idx[e];
        if (hh == 0) dstr[m] = d;
        att[m][hh] = ev[(long)e * NH + hh] / z[d * NH + hh];
      } else {
        if (hh == 0) dstr[m] = 0;
        att[m][hh] = 0.f;
      }
    }
    __syncthreads();
    v8f c = {};
#pragma unroll
    for (int ch = 0; ch < KV_K / 32; ++ch) {
      v16bf a = load_a_frag(A, KV_K, lane, ch);
      c = wmma_bf16(a, bfrag[ch], c);
    }
#pragma unroll
    for (int r = 0; r < 8; ++r) {
      int row = r + hi * 8;
      int e = t * 16 + row;
      if (e < E) {
        float val = (c[r] + bvc) * att[row][head];
        atomicAdd(&agg[(long)dstr[row] * DOUT + col], val);
      }
    }
    __syncthreads();
  }
}

// ---- output GEMM + ReLU + LayerNorm -------------------------------------
__global__ void __launch_bounds__(256)
k_out(const float* __restrict__ h, const float* __restrict__ agg,
      const bf16* __restrict__ wos, const float* __restrict__ bout,
      const float* __restrict__ ln_g, const float* __restrict__ ln_b,
      float* __restrict__ out, int num_dst) {
  __shared__ bf16  A[16 * O_K];
  __shared__ float R[16][DOUT];
  int tile = blockIdx.x, tid = threadIdx.x;
  for (int idx = tid; idx < 16 * O_K; idx += 256) {
    int row = idx / O_K, k = idx - row * O_K;
    int d = tile * 16 + row;
    float v = 0.f;
    if (d < num_dst) {
      if (k < DOUT)       v = agg[(long)d * DOUT + k];
      else if (k < O_KIN) v = h[(long)d * DN + (k - DOUT)];
    }
    A[idx] = (bf16)v;
  }
  __syncthreads();
  int lane = tid & 31, wid = tid >> 5;
  int col = wid * 16 + (lane & 15);
  int hi  = lane >> 4;
  v8f c = {};
#pragma unroll
  for (int ch = 0; ch < O_K / 32; ++ch) {
    v16bf a = load_a_frag(A, O_K, lane, ch);
    v16bf b = load_b_frag(wos, O_K / 32, lane, wid, ch);
    c = wmma_bf16(a, b, c);
  }
  float bc = bout[col];
#pragma unroll
  for (int r = 0; r < 8; ++r) {
    float v = c[r] + bc;
    R[r + hi * 8][col] = v > 0.f ? v : 0.f;   // relu
  }
  __syncthreads();
  if (tid < 16) {
    int d = tile * 16 + tid;
    if (d < num_dst) {
      float s = 0.f, s2 = 0.f;
      for (int n = 0; n < DOUT; ++n) { float v = R[tid][n]; s += v; s2 += v * v; }
      float mean = s * (1.f / DOUT);
      float var  = s2 * (1.f / DOUT) - mean * mean;
      float inv  = rsqrtf(var + 1e-5f);
      for (int n = 0; n < DOUT; ++n)
        out[(long)d * DOUT + n] = (R[tid][n] - mean) * inv * ln_g[n] + ln_b[n];
    }
  }
}

// ---- host launcher ------------------------------------------------------
extern "C" void kernel_launch(void* const* d_in, const int* in_sizes, int n_in,
                              void* d_out, int out_size, void* d_ws, size_t ws_size,
                              hipStream_t stream) {
  const float* h        = (const float*)d_in[0];
  const float* ef       = (const float*)d_in[1];
  const float* dtv      = (const float*)d_in[2];
  const int*   dst_idx  = (const int*)d_in[3];
  // d_in[4] = num_dst scalar on device; derived on host instead.
  const float* time_w   = (const float*)d_in[5];
  const float* time_b   = (const float*)d_in[6];
  const float* wq       = (const float*)d_in[7];
  const float* bq       = (const float*)d_in[8];
  const float* wk       = (const float*)d_in[9];
  const float* bk       = (const float*)d_in[10];
  const float* wv       = (const float*)d_in[11];
  const float* bv       = (const float*)d_in[12];
  const float* att_bias = (const float*)d_in[13];
  const float* wout     = (const float*)d_in[14];
  const float* bout     = (const float*)d_in[15];
  const float* ln_g     = (const float*)d_in[16];
  const float* ln_b     = (const float*)d_in[17];
  float* out = (float*)d_out;

  int E       = in_sizes[3];
  int num_dst = in_sizes[0] / DN - E;

  char* base = (char*)d_ws;
  size_t off = 0;
  auto carve = [&](size_t bytes) {
    char* p = base + off;
    off = (off + bytes + 255) & ~(size_t)255;
    return p;
  };
  float*    q_nodes = (float*)carve((size_t)num_dst * DOUT * 4);
  float*    scores  = (float*)carve((size_t)E * NH * 4);
  float*    ev      = (float*)carve((size_t)E * NH * 4);
  unsigned* m_u     = (unsigned*)carve((size_t)num_dst * NH * 4);
  float*    z       = (float*)carve((size_t)num_dst * NH * 4);
  float*    agg     = (float*)carve((size_t)num_dst * DOUT * 4);
  bf16*     wks     = (bf16*)carve((size_t)KV_K * DOUT * 2);
  bf16*     wvs     = (bf16*)carve((size_t)KV_K * DOUT * 2);
  bf16*     wqs     = (bf16*)carve((size_t)Q_K * DOUT * 2);
  bf16*     wos     = (bf16*)carve((size_t)O_K * DOUT * 2);

  int ntiles = (E + 15) / 16;
  int nblk   = ntiles < 2048 ? ntiles : 2048;

  k_prep_weights<<<(KV_K * DOUT + 255) / 256, 256, 0, stream>>>(
      wq, wk, wv, wout, wqs, wks, wvs, wos);
  k_init<<<(num_dst * DOUT + 255) / 256, 256, 0, stream>>>(m_u, z, agg, num_dst);
  k_qnodes<<<(num_dst + 15) / 16, 256, 0, stream>>>(h, time_b, wqs, bq, q_nodes, num_dst);
  k_kv_score<<<nblk, 256, 0, stream>>>(h, ef, dtv, dst_idx, time_w, time_b,
                                       wks, bk, att_bias, q_nodes, scores, m_u, E, num_dst);
  k_exp<<<(E * NH + 255) / 256, 256, 0, stream>>>(scores, dst_idx, m_u, ev, z, E);
  k_v_agg<<<nblk, 256, 0, stream>>>(h, ef, dtv, dst_idx, time_w, time_b,
                                    wvs, bv, ev, z, agg, E, num_dst);
  k_out<<<(num_dst + 15) / 16, 256, 0, stream>>>(h, agg, wos, bout, ln_g, ln_b, out, num_dst);

  (void)n_in; (void)out_size; (void)ws_size;
}
